// WKV2D_37744172597880
// MI455X (gfx1250) — matrix operand order repaired and checked by reading, not compiled, not run on hardware
//
#include <hip/hip_runtime.h>

typedef __attribute__((ext_vector_type(16))) _Float16 v16h;
typedef __attribute__((ext_vector_type(8)))  float    v8f;
typedef __attribute__((ext_vector_type(4)))  unsigned int v4u;
typedef __attribute__((ext_vector_type(8)))  int     v8i;
typedef __attribute__((ext_vector_type(4)))  int     v4i;

#define Bsz  8
#define Csz  256
#define ND   31      // Manhattan distances 0..30
#define NMT  16      // M-tiles: 256 out positions / 16
#define NKT  8       // K-tiles: 256 src positions / 32
#define CBLK 16      // channels (N) per workgroup
#define NQ   30      // distinct dy-classes: q = (mt - 2*kt) + 14 in [0,29]

// ---------------------------------------------------------------------------
// Setup: Manhattan indicator fragments, de-duplicated by dy-class.
// Fragment (d, q) holds A[m][k] = ( |dyq - (k>>4)| + |m - (k&15)| == d ),
// dyq = q - 14, pre-swizzled into the WMMA f16 A layout (one v16h per lane):
//   lane<16 : row m = lane,    K = {0..7,16..23}
//   lane>=16: row m = lane-16, K = {8..15,24..31}
// Total 31*30 frags * 1KB = ~0.93 MB in d_ws (L2-hot, shared by all WGs).
// ---------------------------------------------------------------------------
__global__ __launch_bounds__(256) void wkv2d_build_g(_Float16* __restrict__ gsw) {
    int gid  = blockIdx.x * 256 + threadIdx.x;
    int lane = gid & 31;
    int fi   = gid >> 5;                 // fragment index
    if (fi >= ND * NQ) return;
    int q   = fi % NQ;
    int d   = fi / NQ;
    int dyq = q - 14;
    int xo  = lane & 15;                 // row m (== out x within tile)
    int kbase = (lane < 16) ? 0 : 8;
    _Float16* dst = gsw + ((size_t)fi * 32 + lane) * 16;
#pragma unroll
    for (int h = 0; h < 16; ++h) {
        int kk = ((h < 8) ? h : (h + 8)) + kbase;
        int dy = dyq - (kk >> 4);        // kk>>4 == (h>=8)
        if (dy < 0) dy = -dy;
        int dx = xo - (kk & 15);
        if (dx < 0) dx = -dx;
        dst[h] = (dy + dx == d) ? (_Float16)1.0f : (_Float16)0.0f;
    }
}

// ---------------------------------------------------------------------------
// Main: WG = (batch b, 16-channel block cb).
//   k_adj (256 out x 16 ch) = sum_d  G_d (256x256) @ T_d (256x16)
//   out = exp(k_adj) * v
// 8 waves x 2 M-tiles; all-zero G fragments skipped with SCALAR branches:
// per-ktile activity precomputed as bitmasks over d (active iff bit0 set,
// mask >>= 1 each iteration) -> ~3 loop-carried SALU per test, EXEC stays
// all-ones at every WMMA.
// k slice staged into LDS by the Tensor Data Mover with row padding 16+1 DW.
// ---------------------------------------------------------------------------
__global__ __launch_bounds__(256) void wkv2d_main(
    const float* __restrict__ w, const float* __restrict__ u,
    const float* __restrict__ k, const float* __restrict__ v,
    const v16h* __restrict__ gsw, float* __restrict__ out)
{
    __shared__ float ksrc[256][17];                       // TDM pads 16 DW + 1 DW
    __shared__ __align__(32) _Float16 tbuf[NKT][32][24];  // B-frags, padded stride

    const int tid  = threadIdx.x;
    const int lane = tid & 31;
    // wave id is uniform within a wave: pin it to an SGPR so all fragment-skip
    // logic and A-fragment addressing is scalar (s_cmp/s_cbranch, no saveexec).
    const int wave = __builtin_amdgcn_readfirstlane(tid >> 5);
    const int b    = blockIdx.x >> 4;
    const int cb   = blockIdx.x & 15;

    // --- TDM: load k[b, :, :, cb*16 .. +15] -> ksrc[256][17] (wave 0 issues) ---
    const float* kb = k + ((size_t)b * 256) * 256 + cb * 16;
    if (wave == 0) {
        unsigned long long ga = (unsigned long long)(uintptr_t)kb;
        unsigned int lds_off  = (unsigned int)(uintptr_t)&ksrc[0][0];
        v4u g0;
        g0.x = 1u;                                        // count=1 (valid), user D#
        g0.y = lds_off;                                   // [63:32]  lds_addr
        g0.z = (unsigned int)(ga & 0xffffffffu);          // [95:64]  global_addr lo
        g0.w = (unsigned int)((ga >> 32) & 0x1ffffffu)    // [120:96] global_addr hi
             | (2u << 30);                                // [127:126] type=2 (image)
        v8i g1;
        g1[0] = (2 << 16)      // data_size = 4 bytes
              | (1 << 20)      // pad_enable
              | (3 << 22)      // pad_interval: 16 DWORDs
              | (0 << 25);     // pad_amount: 1 DWORD  -> row stride 17 floats
        g1[1] = (16 << 16);    // tensor_dim0 = 16 elements   ([63:48] low bits)
        g1[2] = (int)(256u << 16); // tensor_dim1 = 256 rows  ([95:80] low bits)
        g1[3] = (16 << 16);    // tile_dim0 = 16              ([127:112])
        g1[4] = 256;           // tile_dim1 = 256             ([143:128])
        g1[5] = 256;           // tensor_dim0_stride = 256 elements (48b, low 32)
        g1[6] = 0;
        g1[7] = 0;
        v4i g2 = {0, 0, 0, 0};
        v4i g3 = {0, 0, 0, 0};
        v8i g4 = {0, 0, 0, 0, 0, 0, 0, 0};   // extra operand on 6-arg toolchain
        __builtin_amdgcn_tensor_load_to_lds(g0, g1, g2, g3, g4, 0);
        __builtin_amdgcn_s_wait_tensorcnt((short)0);
    }

    const int   chl = tid & 15;
    const float wch = w[cb * 16 + chl];
    const float uch = u[cb * 16 + chl];

    v8f acc0 = {}; v8f acc1 = {};
    const int mt0 = wave * 2;

    // Per-ktile activity bitmasks over d (bit d set iff fragment nonzero at d)
    // and dy-class fragment indices.  All scalar, computed ONCE before the loop.
    unsigned am0[NKT], am1[NKT];
    int q0[NKT];
#pragma unroll
    for (int kt = 0; kt < NKT; ++kt) {
        int dq  = mt0 - 2 * kt;
        int a0  = dq < 0 ? -dq : dq;          // |dq|
        int dm  = dq - 1;
        int am  = dm < 0 ? -dm : dm;          // |dq-1|
        int dp  = dq + 1;
        int ap  = dp < 0 ? -dp : dp;          // |dq+1|
        int lo0 = a0 < am ? a0 : am;          // mtile mt0:  min(|dq|,|dq-1|)
        int lo1 = ap < a0 ? ap : a0;          // mtile mt0+1: min(|dq+1|,|dq|)
        am0[kt] = 0x1FFFFu << lo0;            // active d: [lo0, lo0+16]
        am1[kt] = 0x1FFFFu << lo1;            // active d: [lo1, lo1+16]
        q0[kt]  = dq + 14;                    // fragment class for mt0
    }

    const int ktb = wave;                     // ktile this thread fills in T_d
    const int kb0 = (lane < 16) ? 0 : 16;     // B layout: lane half -> K {0..7,16..23} / {8..15,24..31}

    for (int d = 0; d < ND; ++d) {
        __syncthreads();                      // tbuf consumers of d-1 done; iter 0: TDM done
        {   // Build T_d[src][ch] = relu(k + wc), B-fragment layout
            const float wc = (d == 0) ? uch : -(float)d * wch;
            _Float16* dst = &tbuf[ktb][lane][0];
#pragma unroll
            for (int h = 0; h < 16; ++h) {
                int src = ktb * 32 + (kb0 >> 1) + ((h < 8) ? h : (h + 8));
                float x = ksrc[src][chl] + wc;
                dst[h] = (_Float16)fmaxf(x, 0.0f);
            }
        }
        __syncthreads();

        if (d + 1 < ND) {   // pull next d's fragment classes toward us
            __builtin_prefetch(gsw + ((size_t)((d + 1) * NQ + q0[0]) * 32) + lane, 0, 1);
            __builtin_prefetch(gsw + ((size_t)((d + 1) * NQ + q0[7]) * 32) + lane, 0, 1);
        }
#pragma unroll
        for (int kt = 0; kt < NKT; ++kt) {
            // loop-carried scalar bitmask tests: s_and/s_cmp/s_lshr only
            unsigned a0b = am0[kt] & 1u;
            unsigned a1b = am1[kt] & 1u;
            am0[kt] >>= 1;
            am1[kt] >>= 1;
            if (a0b | a1b) {
                v16h bf = *(const v16h*)&tbuf[kt][lane][0];
                const v16h* gq = gsw + ((size_t)(d * NQ + q0[kt]) * 32) + lane;
                if (a0b) {
                    v16h a0 = gq[0];
                    acc0 = __builtin_amdgcn_wmma_f32_16x16x32_f16(
                               false, a0, false, bf, (short)0, acc0, false, false);
                }
                if (a1b) {
                    v16h a1 = gq[32];   // class q0+1 == mtile mt0+1
                    acc1 = __builtin_amdgcn_wmma_f32_16x16x32_f16(
                               false, a1, false, bf, (short)0, acc1, false, false);
                }
            }
        }
    }

    // Epilogue: C/D layout (VGPR j -> M = j or j+8 by lane half, N = lane%16)
    const int row0 = (lane < 16) ? 0 : 8;
    const int n    = lane & 15;
    const size_t cg = (size_t)cb * 16 + n;
    const float* vb = v   + ((size_t)b * 256) * 256;
    float*       ob = out + ((size_t)b * 256) * 256;
#pragma unroll
    for (int j = 0; j < 8; ++j) {
        int op0 = mt0 * 16 + row0 + j;
        size_t i0 = (size_t)op0 * 256 + cg;
        ob[i0] = __expf(acc0[j]) * vb[i0];
        int op1 = (mt0 + 1) * 16 + row0 + j;
        size_t i1 = (size_t)op1 * 256 + cg;
        ob[i1] = __expf(acc1[j]) * vb[i1];
    }
}

// ---------------------------------------------------------------------------
// Inputs (setup_inputs order): 0:B 1:H 2:W 3:C 4:w[256] 5:u[256]
//                              6:k[8*16*16*256] 7:v[same]   (all f32)
// d_ws: ~0.93 MB for de-duplicated swizzled G fragments (rebuilt every call).
// ---------------------------------------------------------------------------
extern "C" void kernel_launch(void* const* d_in, const int* in_sizes, int n_in,
                              void* d_out, int out_size, void* d_ws, size_t ws_size,
                              hipStream_t stream) {
    const float* w = (const float*)d_in[4];
    const float* u = (const float*)d_in[5];
    const float* k = (const float*)d_in[6];
    const float* v = (const float*)d_in[7];
    float* out = (float*)d_out;
    _Float16* gsw = (_Float16*)d_ws;

    // 31*30 fragments * 32 lanes = 29760 thread-slots
    int nslots = ND * NQ * 32;
    wkv2d_build_g<<<(nslots + 255) / 256, 256, 0, stream>>>(gsw);
    // 8 batches * 16 channel-blocks = 128 WGs of 8 waves
    wkv2d_main<<<Bsz * (Csz / CBLK), 256, 0, stream>>>(
        w, u, k, v, (const v16h*)gsw, out);
}
